// VectorQuantizer_59442347376944
// MI455X (gfx1250) — matrix operand order repaired and checked by reading, not compiled, not run on hardware
//
#include <hip/hip_runtime.h>
#include <hip/hip_bf16.h>

typedef __attribute__((ext_vector_type(16))) _Float16 v16h;
typedef __attribute__((ext_vector_type(8)))  float    v8f;
typedef __attribute__((ext_vector_type(8)))  unsigned vq_v8u;

#define VQ_N      131072   // B*H*W
#define VQ_C      64
#define VQ_K      512
#define VQ_B      32
#define VQ_HW     4096     // H*W
#define VQ_MTILE  128      // rows per workgroup
#define VQ_NWG    1024     // VQ_N / VQ_MTILE (32 tiles per image)
#define XPAD      66       // 33-dword row stride, gcd(33,64)=1 -> conflict-free
#define CBPAD     72       // 144 B row stride: 16B-aligned for async b128 -> LDS

// ---------------- kernel 1: codebook -> f16, norms, zero hist ----------------
__global__ __launch_bounds__(256) void vq_prep_59442347376944(
    const float* __restrict__ cb, _Float16* __restrict__ cbF16,
    float* __restrict__ cbnorm, unsigned int* __restrict__ histG) {
  int t = blockIdx.x * 256 + threadIdx.x;   // 0..511 (grid = 2 blocks)
  float s = 0.f;
  for (int c = 0; c < VQ_C; ++c) {
    float v = cb[t * VQ_C + c];
    s += v * v;
    cbF16[t * VQ_C + c] = (_Float16)v;
  }
  cbnorm[t] = s;
  for (int i = 0; i < 32; ++i) histG[i * VQ_K + t] = 0u;  // 32*512 = 16384
}

// ---------------- kernel 2: WMMA distances + argmin + outputs ----------------
__global__ __launch_bounds__(256) void vq_main_59442347376944(
    const float* __restrict__ x, const float* __restrict__ cbF32,
    const _Float16* __restrict__ cbF16, const float* __restrict__ cbnorm,
    float* __restrict__ outQ, float* __restrict__ partial,
    unsigned int* __restrict__ histG) {
  __shared__ __align__(16) _Float16 sX[VQ_MTILE * XPAD];   // 16896 B
  __shared__ __align__(16) _Float16 sCB[256 * CBPAD];      // 36864 B (half codebook)
  __shared__ float    sNorm[VQ_K];                         // 2048 B
  __shared__ int      sIdx[VQ_MTILE];                      // 512 B
  __shared__ unsigned sHist[VQ_K];                         // 2048 B
  __shared__ float    sRed[8];

  const int tid = threadIdx.x;
  const int wg  = blockIdx.x;
  const int b   = wg >> 5;                 // image index
  const int hwbase = (wg & 31) * VQ_MTILE; // hw offset inside image
  const size_t xbase = (size_t)b * VQ_C * VQ_HW + hwbase;

  for (int i = tid; i < VQ_K; i += 256) { sHist[i] = 0u; sNorm[i] = cbnorm[i]; }

  // stage x tile [p=0..127][c=0..63] as f16 (coalesced global, conflict-free LDS)
  for (int it = 0; it < 32; ++it) {
    int e = it * 256 + tid;
    int c = e >> 7, p = e & 127;
    float v = x[xbase + (size_t)c * VQ_HW + p];
    sX[p * XPAD + c] = (_Float16)v;
  }

  const int  wave    = tid >> 5;
  const int  lane    = tid & 31;
  const int  l15     = lane & 15;
  const bool hiHalf  = lane >= 16;
  const int  rowBase = wave * 16;

  float best[8];
  int   bidx[8];
#pragma unroll
  for (int i = 0; i < 8; ++i) { best[i] = 3.4e38f; bidx[i] = 0; }

  v16h a0, a1;
  bool aLoaded = false;

  for (int half = 0; half < 2; ++half) {
    // ---- async DMA: 256 codes (32 KB) global -> LDS, no VGPR round-trip ----
    // 2048 b128 transfers / 256 threads = 8 per thread; dest rows 144B-aligned.
#pragma unroll
    for (int it = 0; it < 8; ++it) {
      int e = it * 256 + tid;          // 0..2047
      int code = e >> 3, seg = e & 7;  // 8 x 16B segments per 128B code row
      const void* gp = (const void*)(cbF16 + ((half * 256 + code) * VQ_C + seg * 8));
      unsigned ldsAddr = (unsigned)(unsigned long long)&sCB[code * CBPAD + seg * 8];
      asm volatile("global_load_async_to_lds_b128 %0, %1, off"
                   :: "v"(ldsAddr), "v"(gp) : "memory");
    }
    asm volatile("s_wait_asynccnt 0x0" ::: "memory");
    __syncthreads();

    if (!aLoaded) {
      // 16-bit A 16x32 layout: lanes 0-15 K in {0..7,16..23}; lanes 16-31 +8
      const int r = rowBase + l15;
      const _Float16* xr = &sX[r * XPAD];
      vq_v8u ua0, ua1;
#pragma unroll
      for (int v = 0; v < 8; ++v) {
        int klo = (v < 4) ? (v * 2) : (16 + (v - 4) * 2);
        int k = klo + (hiHalf ? 8 : 0);
        ua0[v] = *(const unsigned*)&xr[k];
        ua1[v] = *(const unsigned*)&xr[k + 32];
      }
      a0 = __builtin_bit_cast(v16h, ua0);
      a1 = __builtin_bit_cast(v16h, ua1);
      aLoaded = true;
    }

#pragma unroll 2
    for (int nt = 0; nt < 16; ++nt) {
      // B 32x16 layout: lane n=l15; lanes 0-15 K in {0..7,16..23}; lanes 16-31 +8
      const _Float16* cr = &sCB[(nt * 16 + l15) * CBPAD];
      const int kb = hiHalf ? 8 : 0;
      vq_v8u ub0, ub1;
#pragma unroll
      for (int v = 0; v < 8; ++v) {
        ub0[v] = *(const unsigned*)&cr[v * 2 + (v >= 4 ? 8 : 0) + kb];
        ub1[v] = *(const unsigned*)&cr[v * 2 + (v >= 4 ? 8 : 0) + kb + 32];
      }
      v16h b0 = __builtin_bit_cast(v16h, ub0);
      v16h b1 = __builtin_bit_cast(v16h, ub1);
      v8f cz = {};
      v8f acc = __builtin_amdgcn_wmma_f32_16x16x32_f16(false, a0, false, b0,
                                                       (short)0, cz, false, false);
      acc = __builtin_amdgcn_wmma_f32_16x16x32_f16(false, a1, false, b1,
                                                   (short)0, acc, false, false);
      const int col = half * 256 + nt * 16 + l15;
      const float cn = sNorm[col];
#pragma unroll
      for (int i = 0; i < 8; ++i) {
        float d = cn - 2.0f * acc[i];   // ||e||^2 - 2 x.e (||x||^2 const per row)
        if (d < best[i]) { best[i] = d; bidx[i] = col; }  // strict < => first idx on tie
      }
    }
    __syncthreads();
  }

  // lex-min reduce across the 16 lanes holding one C/D column each
#pragma unroll
  for (int i = 0; i < 8; ++i) {
    float bv = best[i]; int bi = bidx[i];
    for (int s = 1; s < 16; s <<= 1) {
      float ov = __shfl_xor(bv, s, 32);
      int   oi = __shfl_xor(bi, s, 32);
      if (ov < bv || (ov == bv && oi < bi)) { bv = ov; bi = oi; }
    }
    if (l15 == 0) {  // lane 0 -> rows 0..7 of tile, lane 16 -> rows 8..15
      int row = rowBase + (hiHalf ? 8 : 0) + i;
      sIdx[row] = bi;
      atomicAdd(&sHist[bi], 1u);
    }
  }
  __syncthreads();

  // flush per-image histogram (integer atomics: deterministic)
  for (int i = tid; i < VQ_K; i += 256) {
    unsigned v = sHist[i];
    if (v) atomicAdd(&histG[b * VQ_K + i], v);
  }

  // coalesced NCHW output + squared-error partial (exact f32 codebook gather)
  float lsum = 0.f;
  for (int it = 0; it < 32; ++it) {
    int e = it * 256 + tid;
    int c = e >> 7, p = e & 127;
    size_t g = xbase + (size_t)c * VQ_HW + p;
    float q  = cbF32[sIdx[p] * VQ_C + c];
    float xv = x[g];
    outQ[g] = q;
    float df = q - xv;
    lsum += df * df;
  }
#pragma unroll
  for (int s = 16; s >= 1; s >>= 1) lsum += __shfl_xor(lsum, s, 32);
  if (lane == 0) sRed[wave] = lsum;
  __syncthreads();
  if (tid == 0) {
    float t = 0.f;
    for (int w = 0; w < 8; ++w) t += sRed[w];   // fixed order: deterministic
    partial[wg] = t;
  }
}

// ---------------- kernel 3: deterministic final reduce -----------------------
__global__ __launch_bounds__(256) void vq_final_59442347376944(
    const float* __restrict__ partial, const unsigned int* __restrict__ histG,
    float* __restrict__ lossOut, float* __restrict__ histOut) {
  __shared__ float sr[256];
  int tid = threadIdx.x;
  float s = 0.f;
  for (int i = 0; i < 4; ++i) s += partial[i * 256 + tid];
  sr[tid] = s;
  __syncthreads();
  for (int step = 128; step >= 1; step >>= 1) {
    if (tid < step) sr[tid] += sr[tid + step];
    __syncthreads();
  }
  if (tid == 0) lossOut[0] = sr[0] * 1.25f / 8388608.0f;  // (1 + 0.25) * mean
  for (int i = tid; i < VQ_B * VQ_K; i += 256) histOut[i] = (float)histG[i];
}

extern "C" void kernel_launch(void* const* d_in, const int* in_sizes, int n_in,
                              void* d_out, int out_size, void* d_ws, size_t ws_size,
                              hipStream_t stream) {
  const float* x  = (const float*)d_in[0];   // [32,64,64,64]
  const float* cb = (const float*)d_in[1];   // [512,64]

  float* outQ    = (float*)d_out;            // 8388608 quantized (NCHW)
  float* lossOut = outQ + 8388608;           // 1
  float* histOut = lossOut + 1;              // 16384

  char* ws = (char*)d_ws;
  _Float16*     cbF16   = (_Float16*)ws;                      // 65536 B
  float*        cbnorm  = (float*)(ws + 65536);               // 2048 B
  float*        partial = (float*)(ws + 65536 + 2048);        // 4096 B
  unsigned int* histG   = (unsigned int*)(ws + 65536 + 2048 + 4096); // 65536 B

  vq_prep_59442347376944<<<2, 256, 0, stream>>>(cb, cbF16, cbnorm, histG);
  vq_main_59442347376944<<<VQ_NWG, 256, 0, stream>>>(x, cb, cbF16, cbnorm,
                                                     outQ, partial, histG);
  vq_final_59442347376944<<<1, 256, 0, stream>>>(partial, histG, lossOut, histOut);
}
